// Head_54039278518505
// MI455X (gfx1250) — matrix operand order repaired and checked by reading, not compiled, not run on hardware
//
#include <hip/hip_runtime.h>

typedef __attribute__((ext_vector_type(16))) _Float16 v16h;
typedef __attribute__((ext_vector_type(8)))  _Float16 v8h;
typedef __attribute__((ext_vector_type(8)))  float    v8f;

constexpr int Bn = 8;
constexpr int Tn = 2048;
constexpr int Cn = 1024;
constexpr int Hn = 128;

#define WMMA(a, b, c) \
  __builtin_amdgcn_wmma_f32_16x16x32_f16(false, (a), false, (b), (short)0, (c), false, false)

__device__ __forceinline__ v16h cat8(v8h a, v8h b) {
  return __builtin_shufflevector(a, b, 0,1,2,3,4,5,6,7,8,9,10,11,12,13,14,15);
}

// DPP row_ror rotate within the 16-lane row: pure-VALU cross-lane reduction,
// no LDS traffic, no dscnt waits (vs ds_bpermute lowering of __shfl_xor).
template <int CTRL>
__device__ __forceinline__ float dpp_rot(float x) {
  return __builtin_bit_cast(float, __builtin_amdgcn_update_dpp(
      0, __builtin_bit_cast(int, x), CTRL, 0xF, 0xF, true));
}
__device__ __forceinline__ float rowmax16(float x) {
  x = fmaxf(x, dpp_rot<0x121>(x));   // row_ror:1
  x = fmaxf(x, dpp_rot<0x122>(x));   // row_ror:2
  x = fmaxf(x, dpp_rot<0x124>(x));   // row_ror:4
  x = fmaxf(x, dpp_rot<0x128>(x));   // row_ror:8
  return x;
}
__device__ __forceinline__ float rowsum16(float x) {
  x += dpp_rot<0x121>(x);
  x += dpp_rot<0x122>(x);
  x += dpp_rot<0x124>(x);
  x += dpp_rot<0x128>(x);
  return x;
}

// ---------------------------------------------------------------------------
// Stage 1: W (C x H, row-major f32) -> Wt (H x C, f16)  (B-fragment friendly)
// ---------------------------------------------------------------------------
__global__ void wt_transpose(const float* __restrict__ W, _Float16* __restrict__ Wt) {
  int tid = blockIdx.x * blockDim.x + threadIdx.x;
  if (tid >= Hn * Cn) return;
  int h = tid / Cn;
  int c = tid - h * Cn;
  Wt[tid] = (_Float16)W[c * Hn + h];
}

// ---------------------------------------------------------------------------
// Stage 2: projection, one matrix per launch (MAT: 0=Q, 1=K, 2=V).
// One wave computes a 16x128 output tile => 64 accumulator VGPRs, no spills.
// x is read 3x total but is L2-resident (64MB vs 192MB L2).
// ---------------------------------------------------------------------------
template <int MAT>
__global__ void proj_one(const float* __restrict__ x,
                         const _Float16* __restrict__ Wt,
                         _Float16* __restrict__ Out) {
  const int wave = (blockIdx.x * blockDim.x + threadIdx.x) >> 5;
  const int lane = threadIdx.x & 31;
  const int hi   = lane >> 4;
  const int lm   = lane & 15;

  v8f z = {0.f, 0.f, 0.f, 0.f, 0.f, 0.f, 0.f, 0.f};
  v8f acc[8];
#pragma unroll
  for (int i = 0; i < 8; ++i) acc[i] = z;

  const long arow = (long)wave * 16 + lm;  // A-fragment row for this lane

  for (int kc = 0; kc < Cn; kc += 32) {
    // A fragment: x[arow][kc + hi*16 + {0..15}], f32 -> f16
    const float4* p = (const float4*)(x + arow * Cn + kc + hi * 16);
    float4 f0 = p[0], f1 = p[1], f2 = p[2], f3 = p[3];
    float tf[16] = {f0.x, f0.y, f0.z, f0.w, f1.x, f1.y, f1.z, f1.w,
                    f2.x, f2.y, f2.z, f2.w, f3.x, f3.y, f3.z, f3.w};
    v16h a;
#pragma unroll
    for (int i = 0; i < 16; ++i) a[i] = (_Float16)tf[i];

#pragma unroll
    for (int nt = 0; nt < 8; ++nt) {
      const long bof = (long)(nt * 16 + lm) * Cn + kc + hi * 16;
      v16h bw = *(const v16h*)(Wt + bof);
      acc[nt] = WMMA(a, bw, acc[nt]);
    }
  }

  const float qscale = 0.08838834764831845f;  // 1/sqrt(128)
#pragma unroll
  for (int nt = 0; nt < 8; ++nt) {
#pragma unroll
    for (int r = 0; r < 8; ++r) {
      const long t = (long)wave * 16 + r + 8 * hi;   // flat row in [0, B*T)
      const int  h = nt * 16 + lm;
      if (MAT == 0) {
        Out[t * Hn + h] = (_Float16)(acc[nt][r] * qscale);
      } else if (MAT == 1) {
        Out[t * Hn + h] = (_Float16)(acc[nt][r]);
      } else {
        const long bb = t >> 11;                     // / T
        const long tl = t & (Tn - 1);
        Out[bb * (long)Hn * Tn + (long)h * Tn + tl] = (_Float16)(acc[nt][r]);
      }
    }
  }
}

// ---------------------------------------------------------------------------
// One 32-key attention block. MASKED=true only for the single diagonal block.
// ---------------------------------------------------------------------------
template <bool MASKED>
__device__ __forceinline__ void attn_block(
    int k0, int q0, int hi, int lm,
    const v16h aQ[4], _Float16* __restrict__ P,
    const _Float16* __restrict__ Kw, const _Float16* __restrict__ Vt,
    long tbase /* b*Tn */, long vbase /* b*Hn*Tn */,
    v8f accO[8], float m[8], float lsum[8]) {
  const v8f z = {0.f, 0.f, 0.f, 0.f, 0.f, 0.f, 0.f, 0.f};
  const long tk0 = tbase + k0 + lm;

  // S = Q K^T for a 16x32 score tile (two 16x16 WMMA columns).
  v8f S0 = z, S1 = z;
#pragma unroll
  for (int c = 0; c < 4; ++c) {
    v16h b0 = *(const v16h*)(Kw + tk0 * Hn + c * 32 + hi * 16);
    v16h b1 = *(const v16h*)(Kw + (tk0 + 16) * Hn + c * 32 + hi * 16);
    S0 = WMMA(aQ[c], b0, S0);
    S1 = WMMA(aQ[c], b1, S1);
  }

  // Online softmax (rows live across 16-lane halves).
  const int qrb = q0 + 8 * hi;
#pragma unroll
  for (int r = 0; r < 8; ++r) {
    float s0 = S0[r];
    float s1 = S1[r];
    if (MASKED) {
      const int qr = qrb + r;
      if (k0 + lm > qr)      s0 = -1e30f;
      if (k0 + 16 + lm > qr) s1 = -1e30f;
    }
    const float mx    = rowmax16(fmaxf(s0, s1));
    const float mnew  = fmaxf(m[r], mx);
    const float alpha = __expf(m[r] - mnew);
    const float p0 = __expf(s0 - mnew);
    const float p1 = __expf(s1 - mnew);
    const float rs = rowsum16(p0 + p1);
    lsum[r] = lsum[r] * alpha + rs;
    m[r] = mnew;
#pragma unroll
    for (int nt = 0; nt < 8; ++nt) accO[nt][r] *= alpha;
    // stage P (C-layout -> row-major 16x32 f16 in LDS)
    P[(r + 8 * hi) * 32 + lm]      = (_Float16)p0;
    P[(r + 8 * hi) * 32 + 16 + lm] = (_Float16)p1;
  }

  // LDS RAW within the wave: LDS pipe is in-order; wait + compiler fence.
  asm volatile("s_wait_dscnt 0" ::: "memory");

  // Reload P in A-fragment layout
  v8h plo = *(const v8h*)(P + lm * 32 + hi * 16);
  v8h phi = *(const v8h*)(P + lm * 32 + hi * 16 + 8);
  v16h aP = cat8(plo, phi);
  asm volatile("s_wait_dscnt 0" ::: "memory");

  // O += P * V  (V transposed -> contiguous B fragments)
#pragma unroll
  for (int nt = 0; nt < 8; ++nt) {
    const long vof = vbase + (long)(nt * 16 + lm) * Tn + k0 + hi * 16;
    v16h bV = *(const v16h*)(Vt + vof);
    accO[nt] = WMMA(aP, bV, accO[nt]);
  }
}

// ---------------------------------------------------------------------------
// Stage 3: flash attention. One wave per 16-query block.
// __launch_bounds__(128, 1): let the allocator use the full VGPR file so
// accO(64) + aQ(32) + S(16) + in-flight K/V fragments stay spill-free.
// ---------------------------------------------------------------------------
__global__ void __launch_bounds__(128, 1)
flash_attn(const _Float16* __restrict__ Qw,
           const _Float16* __restrict__ Kw,
           const _Float16* __restrict__ Vt,
           float* __restrict__ out) {
  __shared__ _Float16 sP[4][16 * 32];           // wave-private P tiles (1KB each)
  const int wib  = threadIdx.x >> 5;
  const int qb   = blockIdx.x * (blockDim.x >> 5) + wib;
  const int lane = threadIdx.x & 31;
  const int hi   = lane >> 4;
  const int lm   = lane & 15;
  const int b    = (qb * 16) >> 11;             // batch
  const int q0   = (qb * 16) & (Tn - 1);        // query offset within batch
  _Float16* P = sP[wib];

  // Resident Q A-fragments (16 x 128 as 4 chunks of 16x32)
  v16h aQ[4];
  {
    const long trow = (long)qb * 16 + lm;
#pragma unroll
    for (int c = 0; c < 4; ++c) {
      v8h lo = *(const v8h*)(Qw + trow * Hn + c * 32 + hi * 16);
      v8h hh = *(const v8h*)(Qw + trow * Hn + c * 32 + hi * 16 + 8);
      aQ[c] = cat8(lo, hh);
    }
  }

  v8f z = {0.f, 0.f, 0.f, 0.f, 0.f, 0.f, 0.f, 0.f};
  v8f accO[8];
  float m[8], lsum[8];
#pragma unroll
  for (int i = 0; i < 8; ++i) { accO[i] = z; m[i] = -1e30f; lsum[i] = 0.f; }

  const long tbase = (long)b * Tn;
  const long vbase = (long)b * Hn * Tn;
  const int kbdiag = q0 >> 5;                   // the single partially-masked block

  for (int kb = 0; kb < kbdiag; ++kb) {
    const int k0 = kb * 32;
    // Prefetch next key/value block (global_prefetch_b8); kb+1 <= kbdiag is valid.
    __builtin_prefetch((const void*)(Kw + (tbase + k0 + 32 + lm) * Hn), 0, 1);
    __builtin_prefetch((const void*)(Vt + vbase + (long)lm * Tn + k0 + 32), 0, 1);
    attn_block<false>(k0, q0, hi, lm, aQ, P, Kw, Vt, tbase, vbase, accO, m, lsum);
  }
  attn_block<true>(kbdiag * 32, q0, hi, lm, aQ, P, Kw, Vt, tbase, vbase, accO, m, lsum);

  // Epilogue: O / lsum, fp32 out (B,T,H)
#pragma unroll
  for (int r = 0; r < 8; ++r) {
    const float inv = 1.0f / lsum[r];
    const long t = (long)qb * 16 + r + 8 * hi;
#pragma unroll
    for (int nt = 0; nt < 8; ++nt) {
      out[t * Hn + nt * 16 + lm] = accO[nt][r] * inv;
    }
  }
}

// ---------------------------------------------------------------------------
extern "C" void kernel_launch(void* const* d_in, const int* in_sizes, int n_in,
                              void* d_out, int out_size, void* d_ws, size_t ws_size,
                              hipStream_t stream) {
  (void)in_sizes; (void)n_in; (void)out_size; (void)ws_size;
  const float* x  = (const float*)d_in[0];
  const float* Wk = (const float*)d_in[1];
  const float* Wq = (const float*)d_in[2];
  const float* Wv = (const float*)d_in[3];

  _Float16* ws  = (_Float16*)d_ws;
  _Float16* WqT = ws;                         // H*C = 131072 halves
  _Float16* WkT = ws + 131072;
  _Float16* WvT = ws + 262144;
  _Float16* Qw  = ws + 393216;                // B*T*H = 2097152 halves each
  _Float16* Kw  = Qw + 2097152;
  _Float16* Vt  = Kw + 2097152;

  const int n = Hn * Cn;
  wt_transpose<<<(n + 255) / 256, 256, 0, stream>>>(Wq, WqT);
  wt_transpose<<<(n + 255) / 256, 256, 0, stream>>>(Wk, WkT);
  wt_transpose<<<(n + 255) / 256, 256, 0, stream>>>(Wv, WvT);

  // B*T/16 = 1024 tiles, 4 waves per 128-thread block, one matrix per launch
  proj_one<0><<<256, 128, 0, stream>>>(x, WqT, Qw);
  proj_one<1><<<256, 128, 0, stream>>>(x, WkT, Kw);
  proj_one<2><<<256, 128, 0, stream>>>(x, WvT, Vt);

  flash_attn<<<256, 128, 0, stream>>>(Qw, Kw, Vt, (float*)d_out);
}
